// DCM_6055903887670
// MI455X (gfx1250) — compile-verified, matrix operand order checked by reading, not run on hardware
//
#include <hip/hip_runtime.h>

#define NN 8
#define CC 512
#define C4 128
#define HH 60
#define WW 60
#define PP 3600

typedef unsigned short u16;
typedef unsigned int u32;

typedef __attribute__((ext_vector_type(16))) __bf16 v16bf;
typedef __attribute__((ext_vector_type(8)))  float  v8f;

__device__ __forceinline__ u16 f2bf(float f) {
  u32 u = __float_as_uint(f);
  u32 r = u + 0x7FFFu + ((u >> 16) & 1u);   // round-to-nearest-even
  return (u16)(r >> 16);
}
__device__ __forceinline__ float bf2f(u16 u) {
  return __uint_as_float(((u32)u) << 16);
}

// CDNA5 async copy: global -> LDS, 16 bytes per lane, tracked by ASYNCcnt.
__device__ __forceinline__ void async_ld128(u32 lds_byte_off, const u16* g) {
  asm volatile("global_load_async_to_lds_b128 %0, %1, off"
               :: "v"(lds_byte_off), "v"(g) : "memory");
}
__device__ __forceinline__ void wait_async0() {
  asm volatile("s_wait_asynccnt 0" ::: "memory");
}

// One lane's 16x32 bf16 fragment (A row / B^T column strip), CDNA5 layout:
// lane holds row m (lane&15); lane>>4 selects K-half:
// elems 0..7 = K[8*kh .. +7], elems 8..15 = K[16+8*kh .. +7].
__device__ __forceinline__ v16bf ld_frag(const u16* p, int khalf) {
  union { uint4 q[2]; v16bf v; } f;
  f.q[0] = *(const uint4*)(p + (khalf << 3));
  f.q[1] = *(const uint4*)(p + 16 + (khalf << 3));
  return f.v;
}

// ---------------------------------------------------------------------------
// Elementwise / small kernels
// ---------------------------------------------------------------------------
__global__ void k_cast_x(const float* __restrict__ x, u16* __restrict__ cat) {
  long long i = (long long)blockIdx.x * blockDim.x + threadIdx.x;
  if (i >= (long long)NN * CC * PP) return;
  long long n = i / ((long long)CC * PP);
  long long r = i % ((long long)CC * PP);
  cat[n * (4LL * CC * PP) + r] = f2bf(x[i]);
}

__global__ void k_cast_w(const float* __restrict__ w, u16* __restrict__ o, int n) {
  int i = blockIdx.x * blockDim.x + threadIdx.x;
  if (i < n) o[i] = f2bf(w[i]);
}

__global__ void k_pool(const float* __restrict__ x, float* __restrict__ pool, int k) {
  int kk = k * k;
  int total = NN * CC * kk;
  int i = blockIdx.x * blockDim.x + threadIdx.x;
  if (i >= total) return;
  int ij = i % kk;  int nc = i / kk;
  int bi = ij / k, bj = ij % k;
  int bh = HH / k, bw = WW / k;
  const float* src = x + (long long)nc * HH * WW;
  float s = 0.f;
  for (int y = bi * bh; y < (bi + 1) * bh; ++y)
    for (int xx = bj * bw; xx < (bj + 1) * bw; ++xx)
      s += src[y * WW + xx];
  pool[(long long)nc * kk + ij] = s / (float)(bh * bw);
}

__global__ void k_g(const float* __restrict__ pool, const float* __restrict__ w2,
                    const float* __restrict__ s2, const float* __restrict__ b2,
                    float* __restrict__ g, int kk) {
  int total = NN * C4 * kk;
  int i = blockIdx.x * blockDim.x + threadIdx.x;
  if (i >= total) return;
  int ij = i % kk;  int t = i / kk;  int oc = t % C4;  int n = t / C4;
  const float* pn = pool + (long long)n * CC * kk;
  const float* wr = w2 + (long long)oc * CC;
  float s = 0.f;
  for (int c = 0; c < CC; ++c) s += wr[c] * pn[c * kk + ij];
  s = s * s2[oc] + b2[oc];
  g[i] = fmaxf(s, 0.f);
}

__global__ void k_dw(const u16* __restrict__ f, const float* __restrict__ g,
                     u16* __restrict__ o, int k) {
  long long total = (long long)NN * C4 * PP;
  long long i = (long long)blockIdx.x * blockDim.x + threadIdx.x;
  if (i >= total) return;
  int p = (int)(i % PP);  long long nc = i / PP;
  int y = p / WW, x = p % WW;
  int pad = (k - 1) / 2;
  const u16* fn = f + nc * PP;
  const float* gn = g + nc * k * k;
  float acc = 0.f;
  for (int di = 0; di < k; ++di) {
    int yy = y + di - pad;
    if (yy < 0 || yy >= HH) continue;
    for (int dj = 0; dj < k; ++dj) {
      int xx = x + dj - pad;
      if (xx < 0 || xx >= WW) continue;
      acc += bf2f(fn[yy * WW + xx]) * gn[di * k + dj];
    }
  }
  o[i] = f2bf(fmaxf(acc, 0.f));
}

// ---------------------------------------------------------------------------
// WMMA bf16 GEMM: out = relu(scale .* (A[M,K] @ B[K,P]) + bias)
// 256 threads (8 waves). Block tile 64(M) x 128(P), K-tile 64,
// LDS double-buffered; A tile staged via global_load_async_to_lds_b128
// (ASYNCcnt), B tile transposed through registers; wave tile 32x32 =
// 2x2 accumulators, 8 x v_wmma_f32_16x16x32_bf16 per K-tile.
// K must be a multiple of 64.
// ---------------------------------------------------------------------------
__global__ __launch_bounds__(256) void k_gemm(
    const u16* __restrict__ A, const u16* __restrict__ B, long long bBatch,
    const float* __restrict__ scale, const float* __restrict__ bias,
    u16* __restrict__ outBF, float* __restrict__ outF, long long oBatch,
    int M, int K, int P) {
  __shared__ u16 As[2][64 * 64];     // [buf][row][k], row stride 64
  __shared__ u16 Bs[2][128 * 72];    // [buf][pixel][k] (transposed), stride 72

  const int tid   = threadIdx.x;
  const int lane  = tid & 31;
  const int wave  = tid >> 5;
  const int wy    = wave >> 2;       // 0..1 -> 32-row group
  const int wx    = wave & 3;        // 0..3 -> 32-col group
  const int m16   = lane & 15;
  const int khalf = lane >> 4;
  const int m0 = blockIdx.y * 64;
  const int p0 = blockIdx.x * 128;
  const int n  = blockIdx.z;

  const u16* An = A + (long long)m0 * K;
  const u16* Bn = B + (long long)n * bBatch;

  // --- A tile: async global->LDS, 16B chunks. rows arow8+32i, chunk archk ---
  const int archk = tid & 7;          // 16-byte chunk in row (8 per 64-elem row)
  const int arow8 = tid >> 3;         // 0..31
  const u16* aga = An + (long long)arow8 * K + (archk << 3);
  const u32 asOff = (u32)((((arow8 << 6) + (archk << 3)) << 1));
  const u32 asBase0 = (u32)(unsigned long long)(&As[0][0]);
  const u32 asBase1 = (u32)(unsigned long long)(&As[1][0]);

  // --- B tile: 64(K) x 128(P) as 2x2 (K x pixel) register-transposed blocks ---
  const int kprow = tid >> 6;         // 0..3
  const int pp    = tid & 63;         // pixel-pair index (fixed)
  const int p     = p0 + (pp << 1);
  const bool inb  = p < P;
  const int pcl   = inb ? p : (P - 2);             // clamped, always valid
  const u16* bptr = Bn + (long long)(kprow << 1) * P + pcl;

  v8f acc[2][2] = {};
  u32 gb0[8], gb1[8];

  auto issueAsyncA = [&](int k0, int buf) {
    const u32 base = (buf ? asBase1 : asBase0) + asOff;
#pragma unroll
    for (int i = 0; i < 2; ++i)
      async_ld128(base + (u32)(i << 12),                      // +32 rows = 4096B
                  aga + k0 + (long long)(i << 5) * K);
  };

  auto loadRegsB = [&](int k0) {
#pragma unroll
    for (int i = 0; i < 8; ++i) {
      const u16* s = bptr + (long long)(k0 + (i << 3)) * P;
      gb0[i] = *(const u32*)s;
      gb1[i] = *(const u32*)(s + P);
    }
    __builtin_prefetch((const void*)(bptr + (long long)(k0 + 64) * P), 0, 1);
    if (!inb) {
#pragma unroll
      for (int i = 0; i < 8; ++i) { gb0[i] = 0; gb1[i] = 0; }
    }
  };

  auto storeTileB = [&](int buf) {
    u16* bs = Bs[buf];
    const int pbase = (pp << 1) * 72 + (kprow << 1);
#pragma unroll
    for (int i = 0; i < 8; ++i) {
      u32 g0 = gb0[i], g1 = gb1[i];
      u32 d0 = (g0 & 0xFFFFu) | (g1 << 16);          // pixel 2pp,   K pair
      u32 d1 = (g0 >> 16) | (g1 & 0xFFFF0000u);      // pixel 2pp+1, K pair
      *(u32*)&bs[pbase + (i << 3)]      = d0;
      *(u32*)&bs[pbase + 72 + (i << 3)] = d1;
    }
  };

  auto compute = [&](int buf) {
    const u16* as = As[buf];
    const u16* bs = Bs[buf];
    const u16* pa0 = &as[(((wy << 5) + m16) << 6)];
    const u16* pa1 = &as[(((wy << 5) + 16 + m16) << 6)];
    const u16* pb0 = &bs[((wx << 5) + m16) * 72];
    const u16* pb1 = &bs[((wx << 5) + 16 + m16) * 72];
#pragma unroll
    for (int ks = 0; ks < 64; ks += 32) {
      v16bf a0 = ld_frag(pa0 + ks, khalf);
      v16bf a1 = ld_frag(pa1 + ks, khalf);
      v16bf b0 = ld_frag(pb0 + ks, khalf);
      v16bf b1 = ld_frag(pb1 + ks, khalf);
      acc[0][0] = __builtin_amdgcn_wmma_f32_16x16x32_bf16(false, a0, false, b0,
                                                          (short)0, acc[0][0], false, false);
      acc[0][1] = __builtin_amdgcn_wmma_f32_16x16x32_bf16(false, a0, false, b1,
                                                          (short)0, acc[0][1], false, false);
      acc[1][0] = __builtin_amdgcn_wmma_f32_16x16x32_bf16(false, a1, false, b0,
                                                          (short)0, acc[1][0], false, false);
      acc[1][1] = __builtin_amdgcn_wmma_f32_16x16x32_bf16(false, a1, false, b1,
                                                          (short)0, acc[1][1], false, false);
    }
  };

  const int KT = K >> 6;   // #64-wide K tiles
  issueAsyncA(0, 0);
  loadRegsB(0);
  storeTileB(0);
  wait_async0();
  __syncthreads();
  for (int kt = 0; kt < KT; ++kt) {
    const bool hasNext = (kt + 1) < KT;
    if (hasNext) {
      issueAsyncA((kt + 1) << 6, (kt + 1) & 1);  // async A for next tile
      loadRegsB((kt + 1) << 6);                  // B globals in flight
    }
    compute(kt & 1);                             // WMMAs hide the latency
    if (hasNext) {
      storeTileB((kt + 1) & 1);
      wait_async0();                             // A tile landed in LDS
      __syncthreads();
    }
  }

  // ---- fused scale/bias/relu epilogue ----
  const int colb  = p0 + (wx << 5) + m16;
  const int rbase = m0 + (wy << 5) + (khalf << 3);
  if (outF) {
    float* ofn = outF + (long long)n * oBatch;
#pragma unroll
    for (int r = 0; r < 2; ++r)
#pragma unroll
      for (int c = 0; c < 2; ++c) {
        int col = colb + (c << 4);
        if (col >= P) continue;
#pragma unroll
        for (int i = 0; i < 8; ++i) {
          int gr = rbase + (r << 4) + i;
          float v = fmaxf(acc[r][c][i] * scale[gr] + bias[gr], 0.f);
          ofn[(long long)gr * P + col] = v;
        }
      }
  } else {
    u16* obn = outBF + (long long)n * oBatch;
#pragma unroll
    for (int r = 0; r < 2; ++r)
#pragma unroll
      for (int c = 0; c < 2; ++c) {
        int col = colb + (c << 4);
        if (col >= P) continue;
#pragma unroll
        for (int i = 0; i < 8; ++i) {
          int gr = rbase + (r << 4) + i;
          float v = fmaxf(acc[r][c][i] * scale[gr] + bias[gr], 0.f);
          obn[(long long)gr * P + col] = f2bf(v);
        }
      }
  }
}

// ---------------------------------------------------------------------------
// Host-side orchestration
// ---------------------------------------------------------------------------
extern "C" void kernel_launch(void* const* d_in, const int* in_sizes, int n_in,
                              void* d_out, int out_size, void* d_ws, size_t ws_size,
                              hipStream_t stream) {
  (void)in_sizes; (void)n_in; (void)out_size; (void)ws_size;
  const float* x     = (const float*)d_in[0];
  const float* w_out = (const float*)d_in[28];
  const float* s_out = (const float*)d_in[29];
  const float* b_out = (const float*)d_in[30];

  char* ws = (char*)d_ws;
  size_t off = 0;
  auto carve = [&](size_t bytes) -> char* {
    char* pch = ws + off;
    off += (bytes + 255) & ~(size_t)255;
    return pch;
  };
  u16*   cat  = (u16*)carve((size_t)NN * 4 * CC * PP * 2);  // [N,2048,P] bf16
  u16*   fbuf = (u16*)carve((size_t)NN * C4 * PP * 2);      // [N,128,P] bf16
  u16*   obuf = (u16*)carve((size_t)NN * C4 * PP * 2);      // [N,128,P] bf16
  float* pool = (float*)carve((size_t)NN * CC * 25 * 4);    // [N,512,k*k]
  float* gbuf = (float*)carve((size_t)NN * C4 * 25 * 4);    // [N,128,k*k]
  u16*   wA   = (u16*)carve((size_t)CC * 4 * CC * 2);       // bf16 weight staging

  const int PGRID = (PP + 127) / 128;   // 29

  {
    long long tot = (long long)NN * CC * PP;
    k_cast_x<<<dim3((unsigned)((tot + 255) / 256)), 256, 0, stream>>>(x, cat);
  }

  const int ks[3] = {1, 3, 5};
  for (int bi = 0; bi < 3; ++bi) {
    int base = 1 + bi * 9;
    const float* w1 = (const float*)d_in[base + 0];
    const float* s1 = (const float*)d_in[base + 1];
    const float* b1 = (const float*)d_in[base + 2];
    const float* w2 = (const float*)d_in[base + 3];
    const float* s2 = (const float*)d_in[base + 4];
    const float* b2 = (const float*)d_in[base + 5];
    const float* wf = (const float*)d_in[base + 6];
    const float* sf = (const float*)d_in[base + 7];
    const float* bf = (const float*)d_in[base + 8];
    int k = ks[bi], kk = k * k;

    // f = relu(s1*(w1 @ x)+b1) : [N,128,P] bf16
    k_cast_w<<<dim3((C4 * CC + 255) / 256), 256, 0, stream>>>(w1, wA, C4 * CC);
    k_gemm<<<dim3(PGRID, C4 / 64, NN), 256, 0, stream>>>(
        wA, cat, (long long)4 * CC * PP, s1, b1,
        fbuf, nullptr, (long long)C4 * PP, C4, CC, PP);

    // g = relu(s2*(w2 @ pool)+b2) : [N,128,k,k]
    k_pool<<<dim3((NN * CC * kk + 255) / 256), 256, 0, stream>>>(x, pool, k);
    k_g<<<dim3((NN * C4 * kk + 255) / 256), 256, 0, stream>>>(pool, w2, s2, b2, gbuf, kk);

    // o = relu(depthwise(f, g)) : [N,128,P] bf16
    k_dw<<<dim3((unsigned)(((long long)NN * C4 * PP + 255) / 256)), 256, 0, stream>>>(
        fbuf, gbuf, obuf, k);

    // d_k = relu(sf*(wf @ o)+bf) -> concat slot [N, 512*(1+bi).., P]
    k_cast_w<<<dim3((CC * C4 + 255) / 256), 256, 0, stream>>>(wf, wA, CC * C4);
    k_gemm<<<dim3(PGRID, CC / 64, NN), 256, 0, stream>>>(
        wA, obuf, (long long)C4 * PP, sf, bf,
        cat + (long long)CC * (1 + bi) * PP, nullptr, (long long)4 * CC * PP,
        CC, C4, PP);
  }

  // out = relu(s_out*(w_out @ cat)+b_out) : [N,512,P] f32
  k_cast_w<<<dim3((CC * 4 * CC + 255) / 256), 256, 0, stream>>>(w_out, wA, CC * 4 * CC);
  k_gemm<<<dim3(PGRID, CC / 64, NN), 256, 0, stream>>>(
      wA, cat, (long long)4 * CC * PP, s_out, b_out,
      nullptr, (float*)d_out, (long long)CC * PP, CC, 4 * CC, PP);
}
